// DSSnetwork_47802986004719
// MI455X (gfx1250) — compile-verified, matrix operand order checked by reading, not run on hardware
//
#include <hip/hip_runtime.h>
#include <hip/hip_bf16.h>
#include <stdint.h>

// ---------------------------------------------------------------------------
// DSS-GNN on MI455X (gfx1250).
//  * All dense GEMMs (K=128) run on v_wmma_f32_16x16x32_bf16 with fp32 acc.
//  * Activations are converted f32->bf16 once per buffer; weights are packed
//    once into per-lane fragment order, so the WMMA inner loop is just
//    4x global_load_b128 + 1x v_wmma per K-step (matrix-pipe bound).
//  * Sparse aggregation / scatter-means via float4 + global_atomic_add_f32
//    (bandwidth-bound, L2-resident working set).
// ---------------------------------------------------------------------------

typedef __attribute__((ext_vector_type(16))) __bf16 v16bf;
typedef __attribute__((ext_vector_type(8)))  float  v8f;

constexpr int cB = 128, cS = 8, cN0 = 64, cDEG = 16, cL = 3, cD = 128, cT = 10;
constexpr int cN     = cB * cS * cN0;   // 65536 replicated nodes
constexpr int cNSUB  = cB * cS;         // 1024 subgraphs
constexpr int cNORIG = cB * cN0;        // 8192 original nodes
constexpr int cE     = cN * cDEG;       // 1048576 edges
constexpr int cEORIG = cNORIG * cDEG;   // 131072 orig edges

static __device__ __forceinline__ unsigned f2bf_u(float f) {
  unsigned u = __float_as_uint(f);
  u += 0x7FFFu + ((u >> 16) & 1u);      // round-to-nearest-even
  return u >> 16;
}

// ----- bulk f32 -> bf16 conversion (8 elements / thread) --------------------
__global__ __launch_bounds__(256) void cvt_bf16(
    const float* __restrict__ X, uint4* __restrict__ Y, long long n8)
{
  long long t = (long long)blockIdx.x * blockDim.x + threadIdx.x;
  if (t >= n8) return;
  const float4* p = (const float4*)(X + t * 8);
  float4 a = p[0], b = p[1];
  uint4 r;
  r.x = f2bf_u(a.x) | (f2bf_u(a.y) << 16);
  r.y = f2bf_u(a.z) | (f2bf_u(a.w) << 16);
  r.z = f2bf_u(b.x) | (f2bf_u(b.y) << 16);
  r.w = f2bf_u(b.z) | (f2bf_u(b.w) << 16);
  Y[t] = r;
}

// ----- pack W[K x Nc] (f32, row-major) into per-lane B fragments ------------
// Fragment order: chunk t = (kk*2+side)*Nc + n holds 16 bf16:
//   e<8 : K = kk*32 + side*8 + e
//   e>=8: K = kk*32 + 16 + side*8 + (e-8)
__global__ __launch_bounds__(256) void pack_weight(
    const float* __restrict__ W, unsigned short* __restrict__ Wp, int K, int Nc)
{
  int t = blockIdx.x * blockDim.x + threadIdx.x;
  if (t >= (K / 32) * 2 * Nc) return;
  int n = t % Nc;
  int rest = t / Nc;                 // kk*2 + side
  int side = rest & 1, kk = rest >> 1;
  unsigned short* dst = Wp + (size_t)t * 16;
  int kbase = kk * 32 + side * 8;
#pragma unroll
  for (int e = 0; e < 8; ++e) {
    dst[e]     = (unsigned short)f2bf_u(W[(size_t)(kbase + e) * Nc + n]);
    dst[8 + e] = (unsigned short)f2bf_u(W[(size_t)(kbase + 16 + e) * Nc + n]);
  }
}

// ---------------------------------------------------------------------------
// C[M x Nc] = A[M x 128] @ W[128 x Nc] (+bias)(+addend)(relu), bf16 inputs.
// 256 threads = 8 waves; wave w -> 16x16 tile (blockIdx.x*16, blockIdx.y*128+w*16).
// ---------------------------------------------------------------------------
template <bool BIAS, bool ADD, bool RELU>
__global__ __launch_bounds__(256) void gemm_bf16_wmma(
    const unsigned short* __restrict__ Abf,   // M x 128 bf16 row-major
    const unsigned short* __restrict__ Wp,    // packed fragments
    const float* __restrict__ bias, const float* __restrict__ addend,
    float* __restrict__ C, int M, int Nc)
{
  const int wave = threadIdx.x >> 5;
  const int lane = threadIdx.x & 31;
  const int m16  = lane & 15;
  const int side = lane >> 4;
  const int row0 = blockIdx.x << 4;
  const int col0 = (blockIdx.y << 7) + (wave << 4);
  if (col0 >= Nc) return;
  const int n = col0 + m16;
  const uint4* __restrict__ arow =
      (const uint4*)(Abf + (size_t)(row0 + m16) * cD);  // 16 uint4 per row

  v8f acc = {};
#pragma unroll
  for (int kk = 0; kk < cD / 32; ++kk) {
    union { uint4 q[2]; v16bf v; } a, b;
    a.q[0] = arow[kk * 4 + side];        // K = kk*32 + side*8 .. +7
    a.q[1] = arow[kk * 4 + side + 2];    // K = kk*32 + 16 + side*8 .. +7
    const uint4* bp =
        (const uint4*)(Wp + ((size_t)((kk * 2 + side) * Nc + n) * 16));
    b.q[0] = bp[0];
    b.q[1] = bp[1];
    acc = __builtin_amdgcn_wmma_f32_16x16x32_bf16(false, a.v, false, b.v,
                                                  (short)0, acc, false, false);
  }
  float bv = BIAS ? bias[n] : 0.0f;
#pragma unroll
  for (int j = 0; j < 8; ++j) {          // C/D: VGPR j -> M = side*8 + j, N = lane%16
    int r = row0 + side * 8 + j;
    size_t idx = (size_t)r * Nc + n;
    float val = acc[j] + bv;
    if (ADD)  val += addend[idx];
    if (RELU) val = fmaxf(val, 0.0f);
    C[idx] = val;
  }
}

// ----- edge aggregation: agg[dst[e]] += msg[src[e]], D=128, float4 granule --
__global__ __launch_bounds__(256) void scatter_add_edges(
    const float* __restrict__ msg, const int* __restrict__ src,
    const int* __restrict__ dst, float* __restrict__ agg, int nE)
{
  long long t = (long long)blockIdx.x * blockDim.x + threadIdx.x;
  if (t >= (long long)nE * (cD / 4)) return;
  int e  = (int)(t >> 5);
  int c4 = ((int)t & 31) << 2;
  float4 mv = *(const float4*)(msg + (size_t)src[e] * cD + c4);
  float* ap = agg + (size_t)dst[e] * cD + c4;
  atomicAdd(ap + 0, mv.x); atomicAdd(ap + 1, mv.y);
  atomicAdd(ap + 2, mv.z); atomicAdd(ap + 3, mv.w);
}

// ----- scatter_mean stage 1: out[idx[r]] += X[r], cnt[idx[r]] += 1 ----------
__global__ __launch_bounds__(256) void scatter_add_rows(
    const float* __restrict__ X, const int* __restrict__ idx,
    float* __restrict__ out, float* __restrict__ cnt, int Mrows)
{
  long long t = (long long)blockIdx.x * blockDim.x + threadIdx.x;
  if (t >= (long long)Mrows * (cD / 4)) return;
  int r  = (int)(t >> 5);
  int c4 = ((int)t & 31) << 2;
  int o  = idx[r];
  float4 v = *(const float4*)(X + (size_t)r * cD + c4);
  float* op = out + (size_t)o * cD + c4;
  atomicAdd(op + 0, v.x); atomicAdd(op + 1, v.y);
  atomicAdd(op + 2, v.z); atomicAdd(op + 3, v.w);
  if (c4 == 0) atomicAdd(&cnt[o], 1.0f);
}

// ----- scatter_mean stage 2: X[r] /= max(cnt[r], 1) -------------------------
__global__ __launch_bounds__(256) void div_rows(
    float* __restrict__ X, const float* __restrict__ cnt, int Mrows)
{
  long long t = (long long)blockIdx.x * blockDim.x + threadIdx.x;
  if (t >= (long long)Mrows * (cD / 4)) return;
  int r  = (int)(t >> 5);
  int c4 = ((int)t & 31) << 2;
  float d = fmaxf(cnt[r], 1.0f);
  float4* p = (float4*)(X + (size_t)r * cD + c4);
  float4 v = *p;
  v.x /= d; v.y /= d; v.z /= d; v.w /= d;
  *p = v;
}

// ----- batchnorm column stats: st[c] += sum, st[128+c] += sumsq -------------
__global__ __launch_bounds__(256) void bn_partial(
    const float* __restrict__ X, float* __restrict__ st)
{
  int c    = threadIdx.x & (cD - 1);
  int half = threadIdx.x >> 7;
  int base = blockIdx.x * 128;          // 128 rows per block
  float s = 0.0f, ss = 0.0f;
  for (int rr = half; rr < 128; rr += 2) {
    float v = X[(size_t)(base + rr) * cD + c];
    s += v; ss += v * v;
  }
  atomicAdd(&st[c], s);
  atomicAdd(&st[cD + c], ss);
}

// ----- turn (sum,sumsq) into (scale, shift): y = x*scale + shift ------------
__global__ void bn_finalize(float* __restrict__ st, const float* __restrict__ gamma,
                            const float* __restrict__ beta, float invM)
{
  int c = threadIdx.x;
  if (c >= cD) return;
  float mu  = st[c] * invM;
  float var = st[cD + c] * invM - mu * mu;
  float sc  = gamma[c] * rsqrtf(var + 1e-5f);
  st[c]      = sc;
  st[cD + c] = beta[c] - mu * sc;
}

// ----- h = relu(bnA(h1pre) + bnB(h2pre)[node_idx]) --------------------------
__global__ __launch_bounds__(256) void fuse_bn_add_relu(
    const float* __restrict__ h1, const float* __restrict__ h2,
    const float* __restrict__ stA, const float* __restrict__ stB,
    const int* __restrict__ nidx, float* __restrict__ out)
{
  long long t = (long long)blockIdx.x * blockDim.x + threadIdx.x;
  if (t >= (long long)cN * (cD / 4)) return;
  int r  = (int)(t >> 5);
  int c4 = ((int)t & 31) << 2;
  int o  = nidx[r];
  float4 a = *(const float4*)(h1 + (size_t)r * cD + c4);
  float4 b = *(const float4*)(h2 + (size_t)o * cD + c4);
  float4 res;
  res.x = fmaxf(a.x * stA[c4 + 0] + stA[cD + c4 + 0] + b.x * stB[c4 + 0] + stB[cD + c4 + 0], 0.0f);
  res.y = fmaxf(a.y * stA[c4 + 1] + stA[cD + c4 + 1] + b.y * stB[c4 + 1] + stB[cD + c4 + 1], 0.0f);
  res.z = fmaxf(a.z * stA[c4 + 2] + stA[cD + c4 + 2] + b.z * stB[c4 + 2] + stB[cD + c4 + 2], 0.0f);
  res.w = fmaxf(a.w * stA[c4 + 3] + stA[cD + c4 + 3] + b.w * stB[c4 + 3] + stB[cD + c4 + 3], 0.0f);
  *(float4*)(out + (size_t)r * cD + c4) = res;
}

// ----- exclusive cumsums of per-graph counts (B=128, trivial) ---------------
__global__ void build_prefix(const int* __restrict__ nnps, const int* __restrict__ nsub,
                             int* __restrict__ off_nn, int* __restrict__ off_ns)
{
  if (threadIdx.x == 0 && blockIdx.x == 0) {
    int a = 0, b = 0;
    for (int i = 0; i < cB; ++i) {
      off_nn[i] = a; a += nnps[i];
      off_ns[i] = b; b += nsub[i];
    }
  }
}

// ----- node_idx / subgraph_id from inputs -----------------------------------
__global__ void build_idx(const int* __restrict__ batch, const int* __restrict__ sbat,
                          const int* __restrict__ snid, const int* __restrict__ off_nn,
                          const int* __restrict__ off_ns, int* __restrict__ nidx,
                          int* __restrict__ sgid)
{
  int n = blockIdx.x * blockDim.x + threadIdx.x;
  if (n >= cN) return;
  int b = batch[n];
  nidx[n] = off_nn[b] + snid[n];
  sgid[n] = off_ns[b] + sbat[n];
}

// ----- out[b,t] = hidden[b,:] . W2[:,t] + b2[t]  (128 x 10, tiny) -----------
__global__ void final_mlp(const float* __restrict__ hid, const float* __restrict__ W2,
                          const float* __restrict__ b2, float* __restrict__ out)
{
  int b = threadIdx.x;              // 0..127
  int t = blockIdx.x;               // 0..9
  if (b >= cB) return;
  float acc = b2[t];
  for (int k = 0; k < 2 * cD; ++k)
    acc += hid[(size_t)b * (2 * cD) + k] * W2[(size_t)k * cT + t];
  out[(size_t)b * cT + t] = acc;
}

// ---------------------------------------------------------------------------
extern "C" void kernel_launch(void* const* d_in, const int* in_sizes, int n_in,
                              void* d_out, int out_size, void* d_ws, size_t ws_size,
                              hipStream_t stream)
{
  (void)in_sizes; (void)n_in; (void)out_size; (void)ws_size;

  const float* x    = (const float*)d_in[0];
  const int*   ei   = (const int*)d_in[1];
  const int*   oei  = (const int*)d_in[2];
  const int*   batch= (const int*)d_in[3];
  const int*   sbat = (const int*)d_in[4];
  const int*   snid = (const int*)d_in[5];
  const int*   nsub = (const int*)d_in[6];
  const int*   nnps = (const int*)d_in[7];
  const int*   sgb  = (const int*)d_in[8];
  const float* Wenc = (const float*)d_in[9];
  const float* benc = (const float*)d_in[10];
  const float* Wr   = (const float*)d_in[11];
  const float* Wn   = (const float*)d_in[12];
  const float* gb   = (const float*)d_in[13];
  const float* bng  = (const float*)d_in[14];
  const float* bnb  = (const float*)d_in[15];
  const float* Wsr  = (const float*)d_in[16];
  const float* Wsn  = (const float*)d_in[17];
  const float* gsb  = (const float*)d_in[18];
  const float* bnsg = (const float*)d_in[19];
  const float* bnsb = (const float*)d_in[20];
  const float* W1   = (const float*)d_in[21];
  const float* b1   = (const float*)d_in[22];
  const float* W2   = (const float*)d_in[23];
  const float* b2   = (const float*)d_in[24];
  float* out = (float*)d_out;

  // ---- workspace carve-out (256B aligned regions) ----
  char* wsp = (char*)d_ws;
  auto carve = [&](size_t bytes) -> char* {
    char* p = wsp;
    wsp += (bytes + 255) & ~(size_t)255;
    return p;
  };
  float* h     = (float*)carve((size_t)cN * cD * 4);
  float* buf1  = (float*)carve((size_t)cN * cD * 4);
  float* buf2  = (float*)carve((size_t)cN * cD * 4);
  float* xs    = (float*)carve((size_t)cNORIG * cD * 4);
  float* sbuf1 = (float*)carve((size_t)cNORIG * cD * 4);
  float* sbuf2 = (float*)carve((size_t)cNORIG * cD * 4);
  float* hsub  = (float*)carve((size_t)cNSUB * cD * 4);
  float* hgr   = (float*)carve((size_t)cB * cD * 4);
  float* hid   = (float*)carve((size_t)cB * 2 * cD * 4);
  float* cntO  = (float*)carve((size_t)cNORIG * 4);
  float* cntS  = (float*)carve((size_t)cNSUB * 4);
  float* cntG  = (float*)carve((size_t)cB * 4);
  float* stA   = (float*)carve(2 * cD * 4);
  float* stB   = (float*)carve(2 * cD * 4);
  int*   nidx  = (int*)carve((size_t)cN * 4);
  int*   sgid  = (int*)carve((size_t)cN * 4);
  int*   offnn = (int*)carve((size_t)cB * 4);
  int*   offns = (int*)carve((size_t)cB * 4);
  // bf16 activation mirrors
  unsigned short* h_bf   = (unsigned short*)carve((size_t)cN * cD * 2);
  unsigned short* xs_bf  = (unsigned short*)carve((size_t)cNORIG * cD * 2);
  unsigned short* hgr_bf = (unsigned short*)carve((size_t)cB * cD * 2);
  // packed bf16 weights (fragment order)
  unsigned short* WencP = (unsigned short*)carve((size_t)cD * cD * 2);
  unsigned short* WnP   = (unsigned short*)carve((size_t)cL * cD * cD * 2);
  unsigned short* WrP   = (unsigned short*)carve((size_t)cL * cD * cD * 2);
  unsigned short* WsnP  = (unsigned short*)carve((size_t)cL * cD * cD * 2);
  unsigned short* WsrP  = (unsigned short*)carve((size_t)cL * cD * cD * 2);
  unsigned short* W1P   = (unsigned short*)carve((size_t)cD * 2 * cD * 2);

  const int* esrc = ei;            const int* edst = ei + cE;
  const int* osrc = oei;           const int* odst = oei + cEORIG;

  auto blocks = [](long long total) { return (unsigned)((total + 255) / 256); };

  auto gemm = [&](const unsigned short* Abf, const unsigned short* Wp,
                  const float* bias, const float* add, float* C,
                  int M, int Nc, int mode) {
    dim3 g(M / 16, (Nc + 127) / 128);
    if (mode == 0)
      gemm_bf16_wmma<false, false, false><<<g, 256, 0, stream>>>(Abf, Wp, nullptr, nullptr, C, M, Nc);
    else if (mode == 1)
      gemm_bf16_wmma<true, false, false><<<g, 256, 0, stream>>>(Abf, Wp, bias, nullptr, C, M, Nc);
    else if (mode == 2)
      gemm_bf16_wmma<true, true, false><<<g, 256, 0, stream>>>(Abf, Wp, bias, add, C, M, Nc);
    else
      gemm_bf16_wmma<true, false, true><<<g, 256, 0, stream>>>(Abf, Wp, bias, nullptr, C, M, Nc);
  };
  auto cvt = [&](const float* X, unsigned short* Y, long long nelem) {
    cvt_bf16<<<blocks(nelem / 8), 256, 0, stream>>>(X, (uint4*)Y, nelem / 8);
  };
  auto pack = [&](const float* W, unsigned short* Wp, int K, int Nc) {
    pack_weight<<<blocks((K / 32) * 2 * Nc), 256, 0, stream>>>(W, Wp, K, Nc);
  };

  // ---- one-time weight packing (bf16 fragment order) ----
  pack(Wenc, WencP, cD, cD);
  for (int i = 0; i < cL; ++i) {
    size_t wo = (size_t)i * cD * cD;
    pack(Wn  + wo, WnP  + wo, cD, cD);
    pack(Wr  + wo, WrP  + wo, cD, cD);
    pack(Wsn + wo, WsnP + wo, cD, cD);
    pack(Wsr + wo, WsrP + wo, cD, cD);
  }
  pack(W1, W1P, cD, 2 * cD);

  // ---- index precompute ----
  build_prefix<<<1, 32, 0, stream>>>(nnps, nsub, offnn, offns);
  build_idx<<<cN / 256, 256, 0, stream>>>(batch, sbat, snid, offnn, offns, nidx, sgid);

  // ---- feature encoder: h = x @ W_enc + b_enc ----
  cvt(x, h_bf, (long long)cN * cD);
  gemm(h_bf, WencP, benc, nullptr, h, cN, cD, 1);

  const long long tN  = (long long)cN * (cD / 4);
  const long long tNO = (long long)cNORIG * (cD / 4);
  const long long tNS = (long long)cNSUB * (cD / 4);
  const long long tE  = (long long)cE * (cD / 4);
  const long long tEO = (long long)cEORIG * (cD / 4);

  for (int i = 0; i < cL; ++i) {
    const size_t wo = (size_t)i * cD * cD;
    const size_t bo = (size_t)i * cD;

    cvt(h, h_bf, (long long)cN * cD);   // refresh bf16 mirror of h

    // replicated-graph conv: buf1 = h@Wnei ; buf2 = scatter(buf1) ; buf1 = h@Wroot + buf2 + b
    gemm(h_bf, WnP + wo, nullptr, nullptr, buf1, cN, cD, 0);
    hipMemsetAsync(buf2, 0, (size_t)cN * cD * 4, stream);
    scatter_add_edges<<<blocks(tE), 256, 0, stream>>>(buf1, esrc, edst, buf2, cE);
    gemm(h_bf, WrP + wo, gb + bo, buf2, buf1, cN, cD, 2);

    // BN stats for h1pre
    hipMemsetAsync(stA, 0, 2 * cD * 4, stream);
    bn_partial<<<cN / 128, 256, 0, stream>>>(buf1, stA);
    bn_finalize<<<1, cD, 0, stream>>>(stA, bng + bo, bnb + bo, 1.0f / (float)cN);

    // cross-subgraph mean: xs = scatter_mean(h, node_idx)
    hipMemsetAsync(xs, 0, (size_t)cNORIG * cD * 4, stream);
    hipMemsetAsync(cntO, 0, (size_t)cNORIG * 4, stream);
    scatter_add_rows<<<blocks(tN), 256, 0, stream>>>(h, nidx, xs, cntO, cN);
    div_rows<<<blocks(tNO), 256, 0, stream>>>(xs, cntO, cNORIG);
    cvt(xs, xs_bf, (long long)cNORIG * cD);

    // original-graph conv on xs
    gemm(xs_bf, WsnP + wo, nullptr, nullptr, sbuf1, cNORIG, cD, 0);
    hipMemsetAsync(sbuf2, 0, (size_t)cNORIG * cD * 4, stream);
    scatter_add_edges<<<blocks(tEO), 256, 0, stream>>>(sbuf1, osrc, odst, sbuf2, cEORIG);
    gemm(xs_bf, WsrP + wo, gsb + bo, sbuf2, sbuf1, cNORIG, cD, 2);

    // BN stats for h2pre
    hipMemsetAsync(stB, 0, 2 * cD * 4, stream);
    bn_partial<<<cNORIG / 128, 256, 0, stream>>>(sbuf1, stB);
    bn_finalize<<<1, cD, 0, stream>>>(stB, bnsg + bo, bnsb + bo, 1.0f / (float)cNORIG);

    // h = relu(bnA(buf1) + bnB(sbuf1)[node_idx])
    fuse_bn_add_relu<<<blocks(tN), 256, 0, stream>>>(buf1, sbuf1, stA, stB, nidx, h);
  }

  // ---- pooling: subgraph mean then graph mean ----
  hipMemsetAsync(hsub, 0, (size_t)cNSUB * cD * 4, stream);
  hipMemsetAsync(cntS, 0, (size_t)cNSUB * 4, stream);
  scatter_add_rows<<<blocks(tN), 256, 0, stream>>>(h, sgid, hsub, cntS, cN);
  div_rows<<<blocks(tNS), 256, 0, stream>>>(hsub, cntS, cNSUB);

  hipMemsetAsync(hgr, 0, (size_t)cB * cD * 4, stream);
  hipMemsetAsync(cntG, 0, (size_t)cB * 4, stream);
  scatter_add_rows<<<blocks(tNS), 256, 0, stream>>>(hsub, sgb, hgr, cntG, cNSUB);
  div_rows<<<blocks((long long)cB * (cD / 4)), 256, 0, stream>>>(hgr, cntG, cB);

  // ---- MLP head ----
  cvt(hgr, hgr_bf, (long long)cB * cD);
  gemm(hgr_bf, W1P, b1, nullptr, hid, cB, 2 * cD, 3);   // relu(hg@W1+b1), [128,256]
  final_mlp<<<cT, cB, 0, stream>>>(hid, W2, b2, out);   // [128,10]
}
